// BahdanauAttention_63960652972564
// MI455X (gfx1250) — compile-verified, matrix operand order checked by reading, not run on hardware
//
#include <hip/hip_runtime.h>
#include <hip/hip_bf16.h>
#include <math.h>

// ---------------------------------------------------------------------------
// Problem constants (from reference): B=64, S=2048, ENC=512, DEC=512, ATTN=256
// ---------------------------------------------------------------------------
#define BB   64
#define SS   2048
#define ENC  512
#define DEC  512
#define ATTN 256

typedef __attribute__((ext_vector_type(16))) __bf16 v16bf;
typedef __attribute__((ext_vector_type(8)))  float  v8f;

static __device__ __forceinline__ __bf16 f2bf(float f) { return (__bf16)f; }

// ---------------------------------------------------------------------------
// Workspace layout (bytes):
//   [0, 262144)          : packed W_enc bf16 B-fragments  (16 kk * 16 tiles * 32 lanes * 16 halves)
//   [262144, 786432)     : scores  [B, S] f32
//   [786432, 851968)     : proj_dec [B, ATTN] f32
// d_out layout (f32):  context [B*ENC]  then  weights [B*S]
// ---------------------------------------------------------------------------
#define WS_WPACK_OFF   0
#define WS_SCORES_OFF  262144
#define WS_PROJDEC_OFF 786432

// ---------------------------------------------------------------------------
// Kernel 0a: pack W_enc[ENC, ATTN] f32 -> bf16 fragments in WMMA B layout.
// Fragment for (kk, tile): lane l supplies column n = tile*16 + (l&15),
// K values: kb=(l>>4)*8 ; halves[0..7]=K0+kb+j, halves[8..15]=K0+16+kb+j.
// Stored contiguously: 32 bytes per lane -> one v16bf load in the GEMM.
// ---------------------------------------------------------------------------
__global__ void pack_wenc_kernel(const float* __restrict__ Wenc,
                                 __bf16* __restrict__ Wpack) {
  const int kk   = blockIdx.x;   // 0..15  (K block of 32)
  const int tile = blockIdx.y;   // 0..15  (N block of 16)
  const int lane = threadIdx.x;  // 0..31
  const int n  = tile * 16 + (lane & 15);
  const int kb = (lane >> 4) * 8;
  const int k0 = kk * 32;
  v16bf frag;
#pragma unroll
  for (int j = 0; j < 8; ++j) {
    frag[j]     = f2bf(Wenc[(k0 + kb + j) * ATTN + n]);
    frag[8 + j] = f2bf(Wenc[(k0 + 16 + kb + j) * ATTN + n]);
  }
  *(v16bf*)(Wpack + ((((size_t)kk * 16 + tile) * 32 + lane) << 4)) = frag;
}

// ---------------------------------------------------------------------------
// Kernel 0b: proj_dec[b, a] = dec_state[b,:] @ W_dec[:, a]   (64 x 256, tiny)
// ---------------------------------------------------------------------------
__global__ void projdec_kernel(const float* __restrict__ dec,
                               const float* __restrict__ Wdec,
                               float* __restrict__ pd) {
  const int b = blockIdx.x;
  const int a = threadIdx.x;  // 256
  float sum = 0.f;
#pragma unroll 4
  for (int e = 0; e < DEC; ++e)
    sum += dec[b * DEC + e] * Wdec[e * ATTN + a];
  pd[b * ATTN + a] = sum;
}

// ---------------------------------------------------------------------------
// Kernel 1: scores[b, s] = v . tanh(proj_dec[b] + enc[b,s,:] @ W_enc)
// grid = (S/16, B); block = 512 (16 waves). Wave w computes the 16x16 tile
// of (s-rows x attn-cols [16w,16w+16)) accumulating over ENC=512 with
// v_wmma_f32_16x16x32_bf16, then reduces columns weighted by v.
// ---------------------------------------------------------------------------
__global__ void scores_kernel(const float* __restrict__ enc,
                              const float* __restrict__ projdec,
                              const __bf16* __restrict__ Wpack,
                              const float* __restrict__ vvec,
                              const unsigned char* __restrict__ mask,
                              float* __restrict__ scores) {
  const int b    = blockIdx.y;
  const int s0   = blockIdx.x * 16;
  const int wave = threadIdx.x >> 5;   // 0..15 : attn col tile
  const int lane = threadIdx.x & 31;
  const int m    = lane & 15;          // A row within tile
  const int kb   = (lane >> 4) * 8;    // A/B K sub-block select

  const float* __restrict__ arow = enc + ((size_t)b * SS + s0 + m) * ENC;
  const __bf16* __restrict__ bbase = Wpack + (((size_t)wave * 32 + lane) << 4);

  v8f acc = {};
#pragma unroll
  for (int kk = 0; kk < ENC / 32; ++kk) {
    const int k0 = kk * 32;
    // --- A fragment: two contiguous 8-float (32B) loads, convert to bf16 ---
    const float* p0 = arow + k0 + kb;
    const float* p1 = arow + k0 + 16 + kb;
    v16bf afrag;
#pragma unroll
    for (int j = 0; j < 8; ++j) {
      afrag[j]     = f2bf(p0[j]);
      afrag[8 + j] = f2bf(p1[j]);
    }
    // --- B fragment: single contiguous 32B load from packed weights ---
    v16bf bfrag = *(const v16bf*)(bbase + ((size_t)kk * 16 * 32 << 4));
    acc = __builtin_amdgcn_wmma_f32_16x16x32_bf16(
        /*neg_a=*/false, afrag, /*neg_b=*/false, bfrag,
        /*c_mod=*/(short)0, acc, /*reuse_a=*/false, /*reuse_b=*/false);
  }

  // Epilogue: t = tanh(acc + pd[n]) * v[n]; reduce over n for each row m.
  const int n  = wave * 16 + (lane & 15);
  const float pd = projdec[b * ATTN + n];
  const float vn = vvec[n];

  __shared__ float red[16];
  if (threadIdx.x < 16) red[threadIdx.x] = 0.f;
  __syncthreads();

  const int mbase = (lane >> 4) * 8;   // C layout: VGPR r -> row mbase + r
#pragma unroll
  for (int r = 0; r < 8; ++r) {
    float t = tanhf(acc[r] + pd) * vn;
    // reduce across the 16 lanes of this half-wave (xor stays within group)
    t += __shfl_xor(t, 1);
    t += __shfl_xor(t, 2);
    t += __shfl_xor(t, 4);
    t += __shfl_xor(t, 8);
    if ((lane & 15) == 0) atomicAdd(&red[mbase + r], t);  // ds_add_f32
  }
  __syncthreads();

  if (threadIdx.x < 16) {
    const int s = s0 + threadIdx.x;
    float sc = red[threadIdx.x];
    if (!mask[(size_t)b * SS + s]) sc = -1e9f;
    scores[(size_t)b * SS + s] = sc;
  }
}

// ---------------------------------------------------------------------------
// Kernel 2: softmax over S per batch; writes weights into d_out and zeroes
// the context region (re-zeroed every call -> safe under graph replay).
// grid = B, block = 256, each thread owns 8 of the 2048 columns.
// ---------------------------------------------------------------------------
__global__ void softmax_kernel(const float* __restrict__ scores,
                               float* __restrict__ out) {
  const int b   = blockIdx.x;
  const int tid = threadIdx.x;
  __shared__ float sdata[256];

  float local[8];
  float mx = -INFINITY;
#pragma unroll
  for (int i = 0; i < 8; ++i) {
    local[i] = scores[(size_t)b * SS + i * 256 + tid];
    mx = fmaxf(mx, local[i]);
  }
  sdata[tid] = mx;
  __syncthreads();
  for (int s = 128; s > 0; s >>= 1) {
    if (tid < s) sdata[tid] = fmaxf(sdata[tid], sdata[tid + s]);
    __syncthreads();
  }
  mx = sdata[0];
  __syncthreads();

  float sum = 0.f;
#pragma unroll
  for (int i = 0; i < 8; ++i) {
    local[i] = __expf(local[i] - mx);
    sum += local[i];
  }
  sdata[tid] = sum;
  __syncthreads();
  for (int s = 128; s > 0; s >>= 1) {
    if (tid < s) sdata[tid] += sdata[tid + s];
    __syncthreads();
  }
  const float inv = 1.f / sdata[0];

  float* __restrict__ weights = out + (size_t)BB * ENC;
#pragma unroll
  for (int i = 0; i < 8; ++i)
    weights[(size_t)b * SS + i * 256 + tid] = local[i] * inv;

  // zero context region for this batch (kernel 3 accumulates with atomics)
  out[(size_t)b * ENC + tid]       = 0.f;
  out[(size_t)b * ENC + 256 + tid] = 0.f;
}

// ---------------------------------------------------------------------------
// Kernel 3: context[b, e] = sum_s weights[b,s] * enc[b,s,e]
// grid = (16 s-chunks, B), block = 128; each thread owns 4 e's (float4),
// accumulates 128 s values, then global_atomic_add_f32 into context.
// Second (unavoidable) full stream of enc_outputs, fully coalesced.
// ---------------------------------------------------------------------------
__global__ void context_kernel(const float* __restrict__ enc,
                               float* __restrict__ out) {
  const int b     = blockIdx.y;
  const int chunk = blockIdx.x;          // 16 chunks of 128 s-values
  const int e4    = threadIdx.x * 4;     // 128 threads -> 512 e
  const float* __restrict__ weights = out + (size_t)BB * ENC;
  const float* __restrict__ base =
      enc + ((size_t)b * SS + (size_t)chunk * 128) * ENC;

  float4 acc = make_float4(0.f, 0.f, 0.f, 0.f);
  for (int ss = 0; ss < 128; ++ss) {
    const float w = weights[(size_t)b * SS + chunk * 128 + ss];  // scalarized
    const float4 ev = *(const float4*)(base + (size_t)ss * ENC + e4);
    acc.x += w * ev.x;
    acc.y += w * ev.y;
    acc.z += w * ev.z;
    acc.w += w * ev.w;
  }
  float* ctx = out + (size_t)b * ENC + e4;
  atomicAdd(ctx + 0, acc.x);
  atomicAdd(ctx + 1, acc.y);
  atomicAdd(ctx + 2, acc.z);
  atomicAdd(ctx + 3, acc.w);
}

// ---------------------------------------------------------------------------
extern "C" void kernel_launch(void* const* d_in, const int* in_sizes, int n_in,
                              void* d_out, int out_size, void* d_ws, size_t ws_size,
                              hipStream_t stream) {
  const float*         dec   = (const float*)d_in[0];          // [B, DEC]
  const float*         enc   = (const float*)d_in[1];          // [B, S, ENC]
  const unsigned char* mask  = (const unsigned char*)d_in[2];  // [B, S] bool
  const float*         Wenc  = (const float*)d_in[3];          // [ENC, ATTN]
  const float*         Wdec  = (const float*)d_in[4];          // [DEC, ATTN]
  const float*         vvec  = (const float*)d_in[5];          // [ATTN]
  float* out = (float*)d_out;

  char* ws = (char*)d_ws;
  __bf16* Wpack  = (__bf16*)(ws + WS_WPACK_OFF);
  float*  scores = (float*)(ws + WS_SCORES_OFF);
  float*  pd     = (float*)(ws + WS_PROJDEC_OFF);

  // 0a) pack W_enc into bf16 WMMA B-fragments
  pack_wenc_kernel<<<dim3(ENC / 32, ATTN / 16), 32, 0, stream>>>(Wenc, Wpack);
  // 0b) proj_dec
  projdec_kernel<<<BB, ATTN, 0, stream>>>(dec, Wdec, pd);
  // 1) scores via bf16 WMMA (16 waves per block, one 16x16 tile per wave)
  scores_kernel<<<dim3(SS / 16, BB), 512, 0, stream>>>(enc, pd, Wpack, vvec,
                                                       mask, scores);
  // 2) softmax -> weights (d_out) + zero context region
  softmax_kernel<<<BB, 256, 0, stream>>>(scores, out);
  // 3) context accumulation
  context_kernel<<<dim3(16, BB), 128, 0, stream>>>(enc, out);
}
